// LRULayer_28707561406810
// MI455X (gfx1250) — compile-verified, matrix operand order checked by reading, not run on hardware
//
#include <hip/hip_runtime.h>

// ---------------- problem constants ----------------
#define LL 8192   // sequence length (GEMM M)
#define HH 1024   // hidden width
#define NN 1024   // state channels
#define CH_T 256  // scan chunk length (2^8 -> lambda^T via 8 squarings)
#define CH_C (LL / CH_T)  // 32 chunks

typedef __attribute__((ext_vector_type(16))) __bf16 v16bf;
typedef __attribute__((ext_vector_type(8)))  float  v8f;
typedef unsigned short ushort_t;
typedef unsigned int   uint_t;

// ---------------- helpers ----------------
__device__ __forceinline__ ushort_t f32_to_bf16(float f) {
  unsigned u = __float_as_uint(f);
  unsigned r = (u + 0x7FFFu + ((u >> 16) & 1u)) >> 16;  // RNE
  return (ushort_t)r;
}
__device__ __forceinline__ float bf16lo_to_f32(uint_t p) {
  return __uint_as_float(p << 16);
}
__device__ __forceinline__ float bf16hi_to_f32(uint_t p) {
  return __uint_as_float(p & 0xFFFF0000u);
}

// Per-lane fragment gather for WMMA bf16 A/B operands (NT layout).
// ISA 16-bit A 16x32 layout: lane l<16 holds row M=l with K in {0..7,16..23},
// lanes 16..31 hold the same rows with K in {8..15,24..31}. B mirrors with N.
__device__ __forceinline__ v16bf load_frag(const ushort_t* __restrict__ rowptr,
                                           int k0, int hi) {
  union { uint4 q[2]; v16bf v; } u;
  u.q[0] = *reinterpret_cast<const uint4*>(rowptr + k0 + hi * 8);       // K+0..7
  u.q[1] = *reinterpret_cast<const uint4*>(rowptr + k0 + 16 + hi * 8);  // K+16..23
  return u.v;
}

// ---------------- prep kernels ----------------
__global__ void k_conv_x(const float* __restrict__ x, ushort_t* __restrict__ xb,
                         int n) {
  int i = blockIdx.x * blockDim.x + threadIdx.x;
  if (i < n) xb[i] = f32_to_bf16(x[i]);
}

__global__ void k_prep_B(const float* __restrict__ Bre,
                         const float* __restrict__ Bim,
                         const float* __restrict__ gamma_log,
                         ushort_t* __restrict__ bnre,
                         ushort_t* __restrict__ bnim, int n) {
  int i = blockIdx.x * blockDim.x + threadIdx.x;
  if (i < n) {
    int ch = i / HH;
    float g = __expf(gamma_log[ch]);
    bnre[i] = f32_to_bf16(Bre[i] * g);
    bnim[i] = f32_to_bf16(Bim[i] * g);
  }
}

__global__ void k_prep_C(const float* __restrict__ Cre,
                         const float* __restrict__ Cim,
                         ushort_t* __restrict__ cre,
                         ushort_t* __restrict__ cimn, int n) {
  int i = blockIdx.x * blockDim.x + threadIdx.x;
  if (i < n) {
    cre[i]  = f32_to_bf16(Cre[i]);
    cimn[i] = f32_to_bf16(-Cim[i]);  // fold Re() subtraction into a WMMA add
  }
}

__global__ void k_lambda(const float* __restrict__ nu_log,
                         const float* __restrict__ theta_log,
                         float* __restrict__ lamRe, float* __restrict__ lamIm) {
  int i = blockIdx.x * blockDim.x + threadIdx.x;
  if (i < NN) {
    float mag = __expf(-__expf(nu_log[i]));
    float ang = __expf(theta_log[i]);
    lamRe[i] = mag * __cosf(ang);
    lamIm[i] = mag * __sinf(ang);
  }
}

// ------- GEMM 1 (fused re+im): Bu = x @ Bn^T, bf16 in, bf16 out, fp32 acc ---
// Block = 8 waves (wave32). Wave tile 32(M)x32(N), waves 4(M)x2(N):
// block tile 128x64. Per k-step: 2 A frags shared by 4 B frags (re+im),
// 8 WMMAs per 12 b128 loads.
__global__ __launch_bounds__(256) void k_gemm_bu(
    const ushort_t* __restrict__ A, const ushort_t* __restrict__ Bre,
    const ushort_t* __restrict__ Bim, ushort_t* __restrict__ OutRe,
    ushort_t* __restrict__ OutIm) {
  const int lane = threadIdx.x & 31;
  const int wave = threadIdx.x >> 5;
  const int wm   = wave & 3;
  const int wn   = wave >> 2;
  const int r    = lane & 15;
  const int hi   = lane >> 4;
  const int m0   = blockIdx.y * 128 + wm * 32;
  const int n0   = blockIdx.x * 64 + wn * 32;

  v8f zero = {};
  v8f accR[2][2], accI[2][2];
#pragma unroll
  for (int i = 0; i < 2; ++i)
#pragma unroll
    for (int j = 0; j < 2; ++j) { accR[i][j] = zero; accI[i][j] = zero; }

  const ushort_t* arow0 = A + (size_t)(m0 + r) * HH;
  const ushort_t* arow1 = A + (size_t)(m0 + 16 + r) * HH;
  for (int k0 = 0; k0 < HH; k0 += 32) {
    __builtin_prefetch(arow0 + k0 + 128, 0, 1);  // global_prefetch_b8
    v16bf a0 = load_frag(arow0, k0, hi);
    v16bf a1 = load_frag(arow1, k0, hi);
#pragma unroll
    for (int j = 0; j < 2; ++j) {
      const size_t boff = (size_t)(n0 + j * 16 + r) * HH;
      v16bf br = load_frag(Bre + boff, k0, hi);
      v16bf bi = load_frag(Bim + boff, k0, hi);
      accR[0][j] = __builtin_amdgcn_wmma_f32_16x16x32_bf16(
          false, a0, false, br, (short)0, accR[0][j], false, false);
      accR[1][j] = __builtin_amdgcn_wmma_f32_16x16x32_bf16(
          false, a1, false, br, (short)0, accR[1][j], false, false);
      accI[0][j] = __builtin_amdgcn_wmma_f32_16x16x32_bf16(
          false, a0, false, bi, (short)0, accI[0][j], false, false);
      accI[1][j] = __builtin_amdgcn_wmma_f32_16x16x32_bf16(
          false, a1, false, bi, (short)0, accI[1][j], false, false);
    }
  }
  // C/D layout: element e of lane -> row = hi*8 + e, col = lane&15
#pragma unroll
  for (int i = 0; i < 2; ++i)
#pragma unroll
    for (int j = 0; j < 2; ++j)
#pragma unroll
      for (int e = 0; e < 8; ++e) {
        int row = m0 + i * 16 + hi * 8 + e;
        int col = n0 + j * 16 + r;
        size_t idx = (size_t)row * NN + col;
        OutRe[idx] = f32_to_bf16(accR[i][j][e]);
        OutIm[idx] = f32_to_bf16(accI[i][j][e]);
      }
}

// ---------------- scan phase A: per-chunk reduction (2 channels/thread) -----
__global__ void k_scan_chunk(const ushort_t* __restrict__ BuRe,
                             const ushort_t* __restrict__ BuIm,
                             const float* __restrict__ lamRe,
                             const float* __restrict__ lamIm,
                             float* __restrict__ chRe, float* __restrict__ chIm) {
  int t = blockIdx.x * blockDim.x + threadIdx.x;  // channel pair, 0..NN/2-1
  int c = blockIdx.y;                             // chunk
  int n = t * 2;
  float lr0 = lamRe[n],     li0 = lamIm[n];
  float lr1 = lamRe[n + 1], li1 = lamIm[n + 1];
  float br0 = 0.f, bi0 = 0.f, br1 = 0.f, bi1 = 0.f;
  const uint_t* pre = (const uint_t*)BuRe;
  const uint_t* pim = (const uint_t*)BuIm;
  int base = c * CH_T;
  for (int j = 0; j < CH_T; ++j) {
    size_t idx = (size_t)(base + j) * (NN / 2) + t;
    uint_t ur = pre[idx], ui = pim[idx];
    float u0r = bf16lo_to_f32(ur), u1r = bf16hi_to_f32(ur);
    float u0i = bf16lo_to_f32(ui), u1i = bf16hi_to_f32(ui);
    float n0r = fmaf(lr0, br0, fmaf(-li0, bi0, u0r));
    float n0i = fmaf(lr0, bi0, fmaf( li0, br0, u0i));
    float n1r = fmaf(lr1, br1, fmaf(-li1, bi1, u1r));
    float n1i = fmaf(lr1, bi1, fmaf( li1, br1, u1i));
    br0 = n0r; bi0 = n0i; br1 = n1r; bi1 = n1i;
  }
  chRe[c * NN + n]     = br0; chIm[c * NN + n]     = bi0;
  chRe[c * NN + n + 1] = br1; chIm[c * NN + n + 1] = bi1;
}

// ---------------- scan phase B: exclusive scan over chunk aggregates --------
__global__ void k_scan_carry(const float* __restrict__ chRe,
                             const float* __restrict__ chIm,
                             const float* __restrict__ lamRe,
                             const float* __restrict__ lamIm,
                             float* __restrict__ caRe, float* __restrict__ caIm) {
  int n = blockIdx.x * blockDim.x + threadIdx.x;
  float pr = lamRe[n], pi = lamIm[n];
  // lambda^CH_T, CH_T = 256 = 2^8 -> 8 complex squarings
#pragma unroll
  for (int s = 0; s < 8; ++s) {
    float t = pr * pr - pi * pi;
    pi = 2.f * pr * pi;
    pr = t;
  }
  float cr = 0.f, ci = 0.f;
  for (int c = 0; c < CH_C; ++c) {
    caRe[c * NN + n] = cr;
    caIm[c * NN + n] = ci;
    float nr = fmaf(pr, cr, fmaf(-pi, ci, chRe[c * NN + n]));
    float ni = fmaf(pr, ci, fmaf( pi, cr, chIm[c * NN + n]));
    cr = nr; ci = ni;
  }
}

// -------- scan phase C: apply carries, emit bf16 states (2 ch/thread) -------
__global__ void k_scan_apply(const ushort_t* __restrict__ BuRe,
                             const ushort_t* __restrict__ BuIm,
                             const float* __restrict__ lamRe,
                             const float* __restrict__ lamIm,
                             const float* __restrict__ caRe,
                             const float* __restrict__ caIm,
                             ushort_t* __restrict__ SreB,
                             ushort_t* __restrict__ SimB) {
  int t = blockIdx.x * blockDim.x + threadIdx.x;  // channel pair
  int c = blockIdx.y;
  int n = t * 2;
  float lr0 = lamRe[n],     li0 = lamIm[n];
  float lr1 = lamRe[n + 1], li1 = lamIm[n + 1];
  float s0r = caRe[c * NN + n],     s0i = caIm[c * NN + n];
  float s1r = caRe[c * NN + n + 1], s1i = caIm[c * NN + n + 1];
  const uint_t* pre = (const uint_t*)BuRe;
  const uint_t* pim = (const uint_t*)BuIm;
  uint_t* ore = (uint_t*)SreB;
  uint_t* oim = (uint_t*)SimB;
  int base = c * CH_T;
  for (int j = 0; j < CH_T; ++j) {
    size_t idx = (size_t)(base + j) * (NN / 2) + t;
    uint_t ur = pre[idx], ui = pim[idx];
    float u0r = bf16lo_to_f32(ur), u1r = bf16hi_to_f32(ur);
    float u0i = bf16lo_to_f32(ui), u1i = bf16hi_to_f32(ui);
    float n0r = fmaf(lr0, s0r, fmaf(-li0, s0i, u0r));
    float n0i = fmaf(lr0, s0i, fmaf( li0, s0r, u0i));
    float n1r = fmaf(lr1, s1r, fmaf(-li1, s1i, u1r));
    float n1i = fmaf(lr1, s1i, fmaf( li1, s1r, u1i));
    s0r = n0r; s0i = n0i; s1r = n1r; s1i = n1i;
    ore[idx] = (uint_t)f32_to_bf16(s0r) | ((uint_t)f32_to_bf16(s1r) << 16);
    oim[idx] = (uint_t)f32_to_bf16(s0i) | ((uint_t)f32_to_bf16(s1i) << 16);
  }
}

// ------- GEMM 2: y = sre@Cre^T + sim@(-Cim)^T + D*x (fused epilogue) --------
// Wave tile 32x32, dual-A (sre,sim) x dual-B (cre,-cim): 8 WMMAs per
// 16 b128 loads per k-step, single fp32 accumulator set.
__global__ __launch_bounds__(256) void k_gemm_out(
    const ushort_t* __restrict__ Sre, const ushort_t* __restrict__ Sim,
    const ushort_t* __restrict__ Cre, const ushort_t* __restrict__ Cimn,
    const float* __restrict__ X, const float* __restrict__ Dg,
    float* __restrict__ Y) {
  const int lane = threadIdx.x & 31;
  const int wave = threadIdx.x >> 5;
  const int wm   = wave & 3;
  const int wn   = wave >> 2;
  const int r    = lane & 15;
  const int hi   = lane >> 4;
  const int m0   = blockIdx.y * 128 + wm * 32;  // L dim
  const int n0   = blockIdx.x * 64 + wn * 32;   // H dim

  v8f zero = {};
  v8f acc[2][2];
#pragma unroll
  for (int i = 0; i < 2; ++i)
#pragma unroll
    for (int j = 0; j < 2; ++j) acc[i][j] = zero;

  const ushort_t* ar0 = Sre + (size_t)(m0 + r) * NN;
  const ushort_t* ar1 = Sre + (size_t)(m0 + 16 + r) * NN;
  const ushort_t* ai0 = Sim + (size_t)(m0 + r) * NN;
  const ushort_t* ai1 = Sim + (size_t)(m0 + 16 + r) * NN;
  for (int k0 = 0; k0 < NN; k0 += 32) {
    __builtin_prefetch(ar0 + k0 + 128, 0, 1);
    v16bf fr0 = load_frag(ar0, k0, hi);
    v16bf fr1 = load_frag(ar1, k0, hi);
    v16bf fi0 = load_frag(ai0, k0, hi);
    v16bf fi1 = load_frag(ai1, k0, hi);
#pragma unroll
    for (int j = 0; j < 2; ++j) {
      const size_t boff = (size_t)(n0 + j * 16 + r) * NN;
      v16bf br = load_frag(Cre + boff, k0, hi);
      v16bf bi = load_frag(Cimn + boff, k0, hi);
      acc[0][j] = __builtin_amdgcn_wmma_f32_16x16x32_bf16(
          false, fr0, false, br, (short)0, acc[0][j], false, false);
      acc[0][j] = __builtin_amdgcn_wmma_f32_16x16x32_bf16(
          false, fi0, false, bi, (short)0, acc[0][j], false, false);
      acc[1][j] = __builtin_amdgcn_wmma_f32_16x16x32_bf16(
          false, fr1, false, br, (short)0, acc[1][j], false, false);
      acc[1][j] = __builtin_amdgcn_wmma_f32_16x16x32_bf16(
          false, fi1, false, bi, (short)0, acc[1][j], false, false);
    }
  }
#pragma unroll
  for (int i = 0; i < 2; ++i)
#pragma unroll
    for (int j = 0; j < 2; ++j)
#pragma unroll
      for (int e = 0; e < 8; ++e) {
        int row = m0 + i * 16 + hi * 8 + e;
        int col = n0 + j * 16 + r;
        size_t idx = (size_t)row * HH + col;
        Y[idx] = acc[i][j][e] + Dg[col] * X[idx];
      }
}

// ---------------- host launch ----------------
extern "C" void kernel_launch(void* const* d_in, const int* in_sizes, int n_in,
                              void* d_out, int out_size, void* d_ws,
                              size_t ws_size, hipStream_t stream) {
  const float* x         = (const float*)d_in[0];
  const float* nu_log    = (const float*)d_in[1];
  const float* theta_log = (const float*)d_in[2];
  const float* B_re      = (const float*)d_in[3];
  const float* B_im      = (const float*)d_in[4];
  const float* C_re      = (const float*)d_in[5];
  const float* C_im      = (const float*)d_in[6];
  const float* Dvec      = (const float*)d_in[7];
  const float* gamma_log = (const float*)d_in[8];
  float* y = (float*)d_out;

  char* w = (char*)d_ws;
  size_t off = 0;
  auto alloc = [&](size_t bytes) {
    void* p = w + off;
    off = (off + bytes + 255) & ~(size_t)255;
    return p;
  };

  ushort_t* xb    = (ushort_t*)alloc((size_t)LL * HH * 2);
  ushort_t* bnre  = (ushort_t*)alloc((size_t)NN * HH * 2);
  ushort_t* bnim  = (ushort_t*)alloc((size_t)NN * HH * 2);
  ushort_t* cre   = (ushort_t*)alloc((size_t)HH * NN * 2);
  ushort_t* cimn  = (ushort_t*)alloc((size_t)HH * NN * 2);
  float*    lamRe = (float*)alloc((size_t)NN * 4);
  float*    lamIm = (float*)alloc((size_t)NN * 4);
  ushort_t* buReB = (ushort_t*)alloc((size_t)LL * NN * 2);  // bf16 Bu
  ushort_t* buImB = (ushort_t*)alloc((size_t)LL * NN * 2);
  ushort_t* sreB  = (ushort_t*)alloc((size_t)LL * NN * 2);  // bf16 states
  ushort_t* simB  = (ushort_t*)alloc((size_t)LL * NN * 2);
  float*    chRe  = (float*)alloc((size_t)CH_C * NN * 4);
  float*    chIm  = (float*)alloc((size_t)CH_C * NN * 4);
  float*    caRe  = (float*)alloc((size_t)CH_C * NN * 4);
  float*    caIm  = (float*)alloc((size_t)CH_C * NN * 4);

  // prep
  k_conv_x<<<(LL * HH + 255) / 256, 256, 0, stream>>>(x, xb, LL * HH);
  k_prep_B<<<(NN * HH + 255) / 256, 256, 0, stream>>>(B_re, B_im, gamma_log,
                                                      bnre, bnim, NN * HH);
  k_prep_C<<<(HH * NN + 255) / 256, 256, 0, stream>>>(C_re, C_im, cre, cimn,
                                                      HH * NN);
  k_lambda<<<(NN + 255) / 256, 256, 0, stream>>>(nu_log, theta_log, lamRe, lamIm);

  // GEMM 1: fused re+im Bu, bf16 output
  dim3 gemm_grid(NN / 64, LL / 128);
  k_gemm_bu<<<gemm_grid, 256, 0, stream>>>(xb, bnre, bnim, buReB, buImB);

  // chunked complex scan over L (2 channels per thread in A/C phases)
  dim3 scan_grid(NN / 2 / 256, CH_C);
  k_scan_chunk<<<scan_grid, 256, 0, stream>>>(buReB, buImB, lamRe, lamIm,
                                              chRe, chIm);
  k_scan_carry<<<NN / 256, 256, 0, stream>>>(chRe, chIm, lamRe, lamIm,
                                             caRe, caIm);
  k_scan_apply<<<scan_grid, 256, 0, stream>>>(buReB, buImB, lamRe, lamIm,
                                              caRe, caIm, sreB, simB);

  // GEMM 2 + fused D*x epilogue
  dim3 out_grid(HH / 64, LL / 128);
  k_gemm_out<<<out_grid, 256, 0, stream>>>(sreB, simB, cre, cimn, x, Dvec, y);
}
